// SHCSATriton_55860344652359
// MI455X (gfx1250) — matrix operand order repaired and checked by reading, not compiled, not run on hardware
//
#include <hip/hip_runtime.h>
#include <hip/hip_bf16.h>

typedef __attribute__((ext_vector_type(16))) __bf16       v16bf;
typedef __attribute__((ext_vector_type(8)))  __bf16       v8bf;
typedef __attribute__((ext_vector_type(8)))  float        v8f;
typedef __attribute__((ext_vector_type(4)))  unsigned int v4u;

#define T_SEQ   4096
#define C_DIM   1024
#define D_HEAD  256
#define NQKV    768
#define NEG_BIG -1e30f

#if defined(__HIP_DEVICE_COMPILE__) && __has_builtin(__builtin_amdgcn_tensor_load_to_lds)
#define HAVE_TDM 1
#endif

__device__ __forceinline__ unsigned short bfbits(float f) {
    return __builtin_bit_cast(unsigned short, (__bf16)f);
}
__device__ __forceinline__ v8bf load8bf(const unsigned short* p) {
    return __builtin_bit_cast(v8bf, *(const v4u*)p);
}
__device__ __forceinline__ v16bf cat8(v8bf lo, v8bf hi) {
    return __builtin_shufflevector(lo, hi, 0, 1, 2, 3, 4, 5, 6, 7,
                                           8, 9, 10, 11, 12, 13, 14, 15);
}
__device__ __forceinline__ v16bf load16bf(const unsigned short* p) {   // 32B contiguous
    return cat8(load8bf(p), load8bf(p + 8));
}
__device__ __forceinline__ v16bf loadA(const unsigned short* p) {      // A-frag: 16 contiguous @p
    return cat8(load8bf(p), load8bf(p + 8));
}

#ifdef HAVE_TDM
typedef unsigned int u32x4 __attribute__((ext_vector_type(4)));
typedef int          i32x4 __attribute__((ext_vector_type(4)));
typedef int          i32x8 __attribute__((ext_vector_type(8)));

// 2-D TDM tile load, bf16 elements (data_size=1). Dims/strides in elements.
__device__ __forceinline__ void tdm_load_2d(unsigned int lds_addr, const void* gptr,
                                            unsigned int dim0, unsigned int dim1,
                                            unsigned int tile0, unsigned int tile1,
                                            unsigned int stride0) {
    unsigned long long ga = (unsigned long long)(uintptr_t)gptr;
    u32x4 g0;
    g0[0] = 1u;                                               // count=1 (valid, user)
    g0[1] = lds_addr;                                         // LDS byte address
    g0[2] = (unsigned int)(ga & 0xFFFFFFFFu);                 // global_addr[31:0]
    g0[3] = (unsigned int)((ga >> 32) & 0x01FFFFFFu) | (2u << 30); // [56:32] | type=2
    i32x8 g1;
    g1[0] = (int)(1u << 16);                                  // data_size=1 (2 bytes)
    g1[1] = (int)((dim0 & 0xFFFFu) << 16);                    // tensor_dim0 lo
    g1[2] = (int)(((dim0 >> 16) & 0xFFFFu) | ((dim1 & 0xFFFFu) << 16));
    g1[3] = (int)(((dim1 >> 16) & 0xFFFFu) | (tile0 << 16));  // tile_dim0
    g1[4] = (int)(tile1 & 0xFFFFu);                           // tile_dim1 (tile_dim2=0)
    g1[5] = (int)stride0;                                     // tensor_dim0_stride lo
    g1[6] = 0;                                                // stride0 hi, stride1 lo
    g1[7] = 0;
    i32x4 z = {0, 0, 0, 0};
#if defined(__clang_major__) && (__clang_major__ >= 23)
    i32x8 z8 = {0, 0, 0, 0, 0, 0, 0, 0};
    __builtin_amdgcn_tensor_load_to_lds(g0, g1, z, z, z8, 0);
#else
    __builtin_amdgcn_tensor_load_to_lds(g0, g1, z, z, 0);
#endif
}
#endif

// ---------------------------------------------------------------------------
// Pre-pass A: x (fp32) -> bf16 contiguous, 8 elements/thread.
// ---------------------------------------------------------------------------
__global__ void cvt_x_bf16(const float* __restrict__ x,
                           unsigned short* __restrict__ xbf) {
    size_t i = ((size_t)blockIdx.x * blockDim.x + threadIdx.x) * 8;
    v8bf o;
    #pragma unroll
    for (int e = 0; e < 8; ++e) o[e] = (__bf16)x[i + e];
    *(v4u*)(xbf + i) = __builtin_bit_cast(v4u, o);
}

// ---------------------------------------------------------------------------
// Pre-pass B: W (1024x768 fp32) -> Wt (768x1024 bf16), coalesced writes.
// ---------------------------------------------------------------------------
__global__ void cvt_wt_bf16(const float* __restrict__ W,
                            unsigned short* __restrict__ wt) {
    int idx = blockIdx.x * blockDim.x + threadIdx.x;   // idx = n*1024 + k
    int n = idx >> 10;
    int k = idx & 1023;
    wt[idx] = bfbits(W[(size_t)k * NQKV + n]);
}

// ---------------------------------------------------------------------------
// Kernel 1: qkv = x @ W + b via bf16 WMMA; emits q (x 1/16), k, v, v^T.
// ---------------------------------------------------------------------------
__global__ void qkv_gemm_wmma(const unsigned short* __restrict__ xbf,
                              const unsigned short* __restrict__ wt,
                              const float* __restrict__ bias,
                              unsigned short* __restrict__ qb,
                              unsigned short* __restrict__ kb,
                              unsigned short* __restrict__ vb,
                              unsigned short* __restrict__ vt) {
    const int lane  = threadIdx.x;
    const int l16   = lane & 15;
    const int hi    = lane >> 4;
    const int mbase = blockIdx.x * 16;
    const int nbase = blockIdx.y * 16;

    const unsigned short* xr = xbf + (size_t)(mbase + l16) * C_DIM;
    const unsigned short* wr = wt + (size_t)(nbase + l16) * C_DIM;

    v8f acc = {};
    #pragma unroll 4
    for (int kc = 0; kc < C_DIM / 32; ++kc) {
        v16bf a = loadA(xr + kc * 32 + (hi ? 16 : 0));
        v16bf b = load16bf(wr + kc * 32 + (hi ? 16 : 0));
        acc = __builtin_amdgcn_wmma_f32_16x16x32_bf16(false, a, false, b,
                                                      (short)0, acc, false, false);
    }

    unsigned short* dst;
    int   col;
    float mul;
    bool  isv = false;
    if (nbase < D_HEAD)          { dst = qb; col = nbase + l16;              mul = 0.0625f; }
    else if (nbase < 2 * D_HEAD) { dst = kb; col = nbase - D_HEAD + l16;     mul = 1.0f; }
    else                         { dst = vb; col = nbase - 2 * D_HEAD + l16; mul = 1.0f; isv = true; }

    const float bv = bias[nbase + l16];
    #pragma unroll
    for (int r = 0; r < 8; ++r) {
        int m = mbase + r + hi * 8;
        unsigned short bits = bfbits((acc[r] + bv) * mul);
        dst[(size_t)m * D_HEAD + col] = bits;
        if (isv) vt[(size_t)col * T_SEQ + m] = bits;
    }
}

// ---------------------------------------------------------------------------
// Kernel 2: flash attention. TDM double-buffers K / V^T tiles into LDS
// (TENSORcnt pipelined); WMMA operands come from ds_load_b128.
// ---------------------------------------------------------------------------
__global__ void flash_attn_wmma(const unsigned short* __restrict__ qb,
                                const unsigned short* __restrict__ kbm,
                                const unsigned short* __restrict__ vtm,
                                const int* __restrict__ n_padd_p,
                                float* __restrict__ out) {
    extern __shared__ unsigned short smem[];
    unsigned short* kbuf  = smem;                    // 2 x 32*256 bf16 (2 x 16KB)
    unsigned short* vtbuf = smem + 2 * 32 * 256;     // 2 x 256*32 bf16 (2 x 16KB)
    unsigned short* pbuf  = smem + 4 * 32 * 256;     // 16*32 bf16 (1KB)

    const int lane  = threadIdx.x;
    const int l16   = lane & 15;
    const int hi    = lane >> 4;
    const int qbase = blockIdx.x * 16;
    const int npadd = *n_padd_p;

    v16bf qa[8];
    {
        const unsigned short* qr = qb + (size_t)(qbase + l16) * D_HEAD;
        #pragma unroll
        for (int kc = 0; kc < 8; ++kc)
            qa[kc] = loadA(qr + kc * 32 + (hi ? 16 : 0));
    }

    v8f yacc[16] = {};
    float mrun[8], lrun[8];
    #pragma unroll
    for (int r = 0; r < 8; ++r) { mrun[r] = NEG_BIG; lrun[r] = 0.0f; }

    const int kstart = (npadd >> 5) << 5;
    const int kend   = ((qbase + 15) >> 5) << 5;

#ifdef HAVE_TDM
    // Prologue: DMA tile 0 into buffer 0.
    if (kstart <= kend) {
        tdm_load_2d((unsigned int)(uintptr_t)kbuf, kbm + (size_t)kstart * D_HEAD,
                    D_HEAD, T_SEQ, D_HEAD, 32, D_HEAD);         // 32 key rows x 256
        tdm_load_2d((unsigned int)(uintptr_t)vtbuf, vtm + (size_t)kstart,
                    T_SEQ, D_HEAD, 32, D_HEAD, T_SEQ);          // 256 rows x 32 keys
    }
#endif

    int buf = 0;
    for (int kbase = kstart; kbase <= kend; kbase += 32) {
        const bool has_next = (kbase + 32 <= kend);
#ifdef HAVE_TDM
        if (has_next) {   // DMA tile t+1 into the other buffer while computing t
            unsigned int ko = (unsigned int)(uintptr_t)(kbuf + (buf ^ 1) * (32 * 256));
            unsigned int vo = (unsigned int)(uintptr_t)(vtbuf + (buf ^ 1) * (256 * 32));
            tdm_load_2d(ko, kbm + (size_t)(kbase + 32) * D_HEAD,
                        D_HEAD, T_SEQ, D_HEAD, 32, D_HEAD);
            tdm_load_2d(vo, vtm + (size_t)(kbase + 32),
                        T_SEQ, D_HEAD, 32, D_HEAD, T_SEQ);
            __builtin_amdgcn_s_wait_tensorcnt(2);   // in-order: current tile done
        } else {
            __builtin_amdgcn_s_wait_tensorcnt(0);
        }
        const unsigned short* kt = kbuf + buf * (32 * 256);
        const unsigned short* vl = vtbuf + buf * (256 * 32);
#else
        const unsigned short* kt = kbm + (size_t)kbase * D_HEAD;   // row stride 256
        const unsigned short* vl = vtm;                            // indexed below
        if (has_next) {
            __builtin_prefetch(kbm + (size_t)(kbase + 32 + l16) * D_HEAD, 0, 3);
            __builtin_prefetch(vtm + (size_t)(lane * 8) * T_SEQ + kbase + 32, 0, 3);
        }
#endif

        // ---- S = Q @ K^T : two 16x16 f32 tiles over 32 keys
        v8f s0 = {}, s1 = {};
        #pragma unroll
        for (int kc = 0; kc < 8; ++kc) {
            int f0 = kc * 32 + (hi ? 16 : 0);
            v16bf b0 = load16bf(kt + (size_t)l16 * D_HEAD + f0);
            v16bf b1 = load16bf(kt + (size_t)(16 + l16) * D_HEAD + f0);
            s0 = __builtin_amdgcn_wmma_f32_16x16x32_bf16(false, qa[kc], false, b0,
                                                         (short)0, s0, false, false);
            s1 = __builtin_amdgcn_wmma_f32_16x16x32_bf16(false, qa[kc], false, b1,
                                                         (short)0, s1, false, false);
        }

        // ---- mask + online softmax
        const int j0 = kbase + l16;
        const int j1 = kbase + 16 + l16;
        #pragma unroll
        for (int r = 0; r < 8; ++r) {
            int m = qbase + r + hi * 8;
            float v0 = ((j0 > m) || (j0 < npadd)) ? NEG_BIG : s0[r];
            float v1 = ((j1 > m) || (j1 < npadd)) ? NEG_BIG : s1[r];
            float rm = fmaxf(v0, v1);
            #pragma unroll
            for (int sh = 1; sh < 16; sh <<= 1)
                rm = fmaxf(rm, __shfl_xor(rm, sh, 32));
            float mnew  = fmaxf(mrun[r], rm);
            float scale = __expf(mrun[r] - mnew);
            float p0 = __expf(v0 - mnew);
            float p1 = __expf(v1 - mnew);
            float rs = p0 + p1;
            #pragma unroll
            for (int sh = 1; sh < 16; sh <<= 1)
                rs += __shfl_xor(rs, sh, 32);
            lrun[r] = lrun[r] * scale + rs;
            mrun[r] = mnew;
            #pragma unroll
            for (int n = 0; n < 16; ++n) yacc[n][r] *= scale;
            int prow = r + hi * 8;
            pbuf[prow * 32 + l16]      = bfbits(p0);
            pbuf[prow * 32 + 16 + l16] = bfbits(p1);
        }
        __syncthreads();
        v16bf pa = loadA(pbuf + l16 * 32 + (hi ? 16 : 0));
        __syncthreads();

        // ---- Y += P @ V : B operand contiguous along keys
        #pragma unroll
        for (int n = 0; n < 16; ++n) {
#ifdef HAVE_TDM
            v16bf vtile = load16bf(vl + (size_t)(n * 16 + l16) * 32 + (hi ? 16 : 0));
#else
            v16bf vtile = load16bf(vl + (size_t)(n * 16 + l16) * T_SEQ + kbase + (hi ? 16 : 0));
#endif
            yacc[n] = __builtin_amdgcn_wmma_f32_16x16x32_bf16(false, pa, false, vtile,
                                                              (short)0, yacc[n], false, false);
        }
        buf ^= 1;
    }

    // ---- epilogue
    #pragma unroll
    for (int n = 0; n < 16; ++n) {
        #pragma unroll
        for (int r = 0; r < 8; ++r) {
            int m = qbase + r + hi * 8;
            float val = 0.0f;
            if (m >= npadd && lrun[r] > 0.0f) val = yacc[n][r] / lrun[r];
            out[(size_t)m * D_HEAD + n * 16 + l16] = val;
        }
    }
}

// ---------------------------------------------------------------------------
extern "C" void kernel_launch(void* const* d_in, const int* in_sizes, int n_in,
                              void* d_out, int out_size, void* d_ws, size_t ws_size,
                              hipStream_t stream) {
    const float* x    = (const float*)d_in[0];
    const float* W    = (const float*)d_in[1];
    const float* bias = (const float*)d_in[2];
    const int* n_padd = (const int*)d_in[3];
    float* out        = (float*)d_out;

    unsigned short* xbf = (unsigned short*)d_ws;                 // 8 MB
    unsigned short* wt  = xbf + (size_t)T_SEQ * C_DIM;           // 1.5 MB
    unsigned short* qb  = wt + (size_t)NQKV * C_DIM;             // 2 MB
    unsigned short* kb  = qb + (size_t)T_SEQ * D_HEAD;           // 2 MB
    unsigned short* vb  = kb + (size_t)T_SEQ * D_HEAD;           // 2 MB
    unsigned short* vt  = vb + (size_t)T_SEQ * D_HEAD;           // 2 MB

    cvt_x_bf16<<<(T_SEQ * C_DIM / 8) / 256, 256, 0, stream>>>(x, xbf);
    cvt_wt_bf16<<<(NQKV * C_DIM) / 256, 256, 0, stream>>>(W, wt);

    dim3 g1(T_SEQ / 16, NQKV / 16);
    qkv_gemm_wmma<<<g1, 32, 0, stream>>>(xbf, wt, bias, qb, kb, vb, vt);

    // dynamic LDS: 2x16KB K tiles + 2x16KB V^T tiles + 1KB P tile = 66,560 B
    flash_attn_wmma<<<T_SEQ / 16, 32, 66560, stream>>>(qb, kb, vt, n_padd, out);
}
